// MistralAttention_86234353369193
// MI455X (gfx1250) — compile-verified, matrix operand order checked by reading, not run on hardware
//
#include <hip/hip_runtime.h>

// ---------------------------------------------------------------------------
// Mistral attention block for MI455X (gfx1250): bf16 WMMA everywhere.
// B=2, S=2048, D=4096, H=32, KVH=8, HD=128.
// ---------------------------------------------------------------------------

typedef __attribute__((ext_vector_type(16))) __bf16 v16bf;
typedef __attribute__((ext_vector_type(8)))  float  v8f;
typedef __attribute__((ext_vector_type(2)))  __bf16 v2bf;

#define BATCH 2
#define SEQ   2048
#define DMODEL 4096
#define NHEADS 32
#define NKVH   8
#define NREP   4
#define HDIM   128
#define ROWS   (BATCH * SEQ)      // 4096
#define QCOLS  (NHEADS * HDIM)    // 4096
#define KVCOLS (NKVH * HDIM)      // 1024
#define ATTN_SCALE 0.08838834764831845f
#define NEG_BIG (-1.0e9f)

// gfx1250 async copy to LDS (ASYNCcnt path), guarded for toolchain support.
// Builtin signature (from hipcc diagnostic): param0 = int4 addrspace(1)*,
// param1 = int4 addrspace(3)*, then two int immediates (offset, cpol).
#if __has_builtin(__builtin_amdgcn_global_load_async_to_lds_b128) && \
    __has_builtin(__builtin_amdgcn_s_wait_asynccnt)
#define USE_ASYNC_LDS 1
typedef int i4v __attribute__((vector_size(16)));
typedef __attribute__((address_space(1))) i4v gi4_t;
typedef __attribute__((address_space(3))) i4v si4_t;
#endif

struct alignas(16) BF8v { __bf16 v[8]; };

__device__ inline __bf16 to_bf16(float x)  { return (__bf16)x; }

// Packed f32x2 -> bf16x2 convert (v_cvt_pk_bf16_f32 when available).
__device__ inline void store4_bf16(__bf16* dst, float4 f) {
#if __has_builtin(__builtin_amdgcn_cvt_pk_bf16_f32)
  ((v2bf*)dst)[0] = __builtin_amdgcn_cvt_pk_bf16_f32(f.x, f.y);
  ((v2bf*)dst)[1] = __builtin_amdgcn_cvt_pk_bf16_f32(f.z, f.w);
#else
  dst[0] = (__bf16)f.x; dst[1] = (__bf16)f.y;
  dst[2] = (__bf16)f.z; dst[3] = (__bf16)f.w;
#endif
}

template <typename T> __device__ inline T from_f32(float x);
template <> __device__ inline float  from_f32<float>(float x)  { return x; }
template <> __device__ inline __bf16 from_f32<__bf16>(float x) { return (__bf16)x; }

// Load a 16x32 bf16 fragment (A layout, or B from [n][k]-major storage).
// ISA 7.12.2: lane l -> row m = l&15, half = l>>4; element e -> k = 8*half + e
// for e<8, 8*half + 16 + (e-8) for e>=8.
__device__ inline v16bf load_frag16(const __bf16* __restrict__ base, int ld) {
  const int lane = threadIdx.x & 31;
  const int m    = lane & 15;
  const int half = lane >> 4;
  const __bf16* row = base + (size_t)m * ld + 8 * half;
  v16bf f;
#pragma unroll
  for (int e = 0; e < 8; ++e) {
    f[e]     = row[e];
    f[e + 8] = row[e + 16];
  }
  return f;
}

__device__ inline v8f wmma_bf16(v16bf a, v16bf b, v8f c) {
  return __builtin_amdgcn_wmma_f32_16x16x32_bf16(
      /*neg_a=*/false, a, /*neg_b=*/false, b,
      /*c_mod=*/(short)0, c, /*reuse_a=*/false, /*reuse_b=*/false);
}

// ---------------------------------------------------------------------------
// Generic WMMA GEMM: C[M,N] = A[M,K] * Bw[K,N].  A: f32 or bf16, Bw: f32.
// BM=BN=128, BK=32; 256 threads = 8 waves, each wave owns a 32x64 sub-tile.
// ---------------------------------------------------------------------------
#define BM 128
#define BN 128
#define BK 32
#define LDA (BK + 8)   // 40 elems, 80B rows (16B aligned)
#define LDB (BK + 8)

template <typename TA, typename TC>
__global__ __launch_bounds__(256) void gemm_wmma_kernel(
    const TA* __restrict__ A, const float* __restrict__ Bw,
    TC* __restrict__ C, int M, int N, int K) {
  __shared__ alignas(16) __bf16 As[BM][LDA];
  __shared__ alignas(16) __bf16 Bs[BN][LDB];   // stored [n][k]

  const int tid  = threadIdx.x;
  const int lane = tid & 31;
  const int wave = tid >> 5;
  const int wm   = wave & 3;    // 0..3 -> 32-row stripe
  const int wn   = wave >> 2;   // 0..1 -> 64-col stripe
  const int m0   = blockIdx.y * BM;
  const int n0   = blockIdx.x * BN;

  v8f acc[2][4];
#pragma unroll
  for (int i = 0; i < 2; ++i)
#pragma unroll
    for (int j = 0; j < 4; ++j)
#pragma unroll
      for (int e = 0; e < 8; ++e) acc[i][j][e] = 0.0f;

  for (int k0 = 0; k0 < K; k0 += BK) {
    // One prefetch hint per operand for the next K-slice (global_prefetch_b8).
    if (k0 + BK < K) {
      __builtin_prefetch(
          &A[(size_t)(m0 + (tid >> 3)) * K + (k0 + BK + (tid & 7) * 4)], 0, 1);
      __builtin_prefetch(
          &Bw[(size_t)(k0 + BK + (tid >> 5)) * N + (n0 + (tid & 31) * 4)], 0, 1);
    }
    // Stage A tile (128x32) -> bf16 LDS. Vectorized 16B loads, 8B packed stores.
    if constexpr (sizeof(TA) == 4) {
      const int c4 = (tid & 7) * 4;
      const int r0 = tid >> 3;          // 0..31
#pragma unroll
      for (int i = 0; i < 4; ++i) {
        const int r = r0 + 32 * i;
        const float4 f = *reinterpret_cast<const float4*>(
            &A[(size_t)(m0 + r) * K + (k0 + c4)]);
        store4_bf16(&As[r][c4], f);
      }
    } else {
      const int c8 = (tid & 3) * 8;
      const int r0 = tid >> 2;          // 0..63
#pragma unroll
      for (int i = 0; i < 2; ++i) {
        const int r = r0 + 64 * i;
        *(BF8v*)&As[r][c8] =
            *reinterpret_cast<const BF8v*>(&A[(size_t)(m0 + r) * K + (k0 + c8)]);
      }
    }
    // Stage B tile (32x128) transposed into Bs[n][k]. 16B global loads.
    {
      const int n4 = (tid & 31) * 4;
      const int kr = tid >> 5;          // 0..7
#pragma unroll
      for (int i = 0; i < 4; ++i) {
        const int k = kr + 8 * i;
        const float4 f = *reinterpret_cast<const float4*>(
            &Bw[(size_t)(k0 + k) * N + (n0 + n4)]);
        Bs[n4 + 0][k] = to_bf16(f.x);
        Bs[n4 + 1][k] = to_bf16(f.y);
        Bs[n4 + 2][k] = to_bf16(f.z);
        Bs[n4 + 3][k] = to_bf16(f.w);
      }
    }
    __syncthreads();

    const v16bf a0 = load_frag16(&As[wm * 32 +  0][0], LDA);
    const v16bf a1 = load_frag16(&As[wm * 32 + 16][0], LDA);
#pragma unroll
    for (int nt = 0; nt < 4; ++nt) {
      const v16bf b = load_frag16(&Bs[wn * 64 + nt * 16][0], LDB);
      acc[0][nt] = wmma_bf16(a0, b, acc[0][nt]);
      acc[1][nt] = wmma_bf16(a1, b, acc[1][nt]);
    }
    __syncthreads();
  }

  // Store: C/D layout — lane = n + 16*(m>=8), VGPR r holds m = r + 8*half.
  const int nl   = lane & 15;
  const int half = lane >> 4;
#pragma unroll
  for (int i = 0; i < 2; ++i)
#pragma unroll
    for (int nt = 0; nt < 4; ++nt)
#pragma unroll
      for (int r = 0; r < 8; ++r) {
        const int row = m0 + wm * 32 + i * 16 + r + 8 * half;
        const int col = n0 + wn * 64 + nt * 16 + nl;
        C[(size_t)row * N + col] = from_f32<TC>(acc[i][nt][r]);
      }
}

// ---------------------------------------------------------------------------
// RoPE (in place, bf16): out[d] = x[d]*c[d] - x[d+64]*s[d];
//                        out[d+64] = x[d+64]*c[d+64] + x[d]*s[d+64]
// ---------------------------------------------------------------------------
__global__ __launch_bounds__(256) void rope_kernel(
    __bf16* __restrict__ X, const float* __restrict__ cosp,
    const float* __restrict__ sinp, int nh) {
  const long long idx   = (long long)blockIdx.x * blockDim.x + threadIdx.x;
  const long long total = (long long)ROWS * nh * (HDIM / 2);
  if (idx >= total) return;
  const int d   = (int)(idx & 63);
  const int h   = (int)((idx >> 6) % nh);
  const int row = (int)(idx / ((long long)nh * 64));
  const size_t base = (size_t)row * nh * HDIM + (size_t)h * HDIM;
  const float x1 = (float)X[base + d];
  const float x2 = (float)X[base + d + 64];
  const float c1 = cosp[(size_t)row * HDIM + d];
  const float c2 = cosp[(size_t)row * HDIM + d + 64];
  const float s1 = sinp[(size_t)row * HDIM + d];
  const float s2 = sinp[(size_t)row * HDIM + d + 64];
  X[base + d]      = (__bf16)(x1 * c1 - x2 * s1);
  X[base + d + 64] = (__bf16)(x2 * c2 + x1 * s2);
}

// ---------------------------------------------------------------------------
// Flash attention (causal, GQA). Block = 128 q rows x one (b,h); 8 waves,
// each wave owns 16 q rows. KV tiles of 64 keys. K tile staged with async
// copy to LDS (ASYNCcnt) when the toolchain exposes it.
// ---------------------------------------------------------------------------
#define LDK (HDIM + 8)   // 136 -> 272B rows (16B aligned)
#define LDV (64 + 8)     // 72  -> 144B rows (16B aligned)
#define LDP (64 + 8)

__global__ __launch_bounds__(256) void flash_attn_kernel(
    const __bf16* __restrict__ Q, const __bf16* __restrict__ Kc,
    const __bf16* __restrict__ Vc, __bf16* __restrict__ O) {
  __shared__ alignas(16) __bf16 Ks[64][LDK];        // [key][d]
  __shared__ alignas(16) __bf16 Vs[HDIM][LDV];      // [d][key] (transposed)
  __shared__ alignas(16) __bf16 Ps[8][16][LDP];     // wave-private P

  const int tid  = threadIdx.x;
  const int lane = tid & 31;
  const int wave = tid >> 5;
  const int nl   = lane & 15;
  const int half = lane >> 4;

  const int q0  = blockIdx.x * 128;
  const int bh  = blockIdx.y;
  const int b   = bh / NHEADS;
  const int h   = bh % NHEADS;
  const int kvh = h / NREP;

  // Preload this wave's 16x128 Q stripe as 4 A-fragments (global bf16).
  const __bf16* qbase =
      Q + (size_t)(b * SEQ + q0 + wave * 16) * QCOLS + (size_t)h * HDIM;
  v16bf aq[4];
#pragma unroll
  for (int ks = 0; ks < 4; ++ks) aq[ks] = load_frag16(qbase + ks * 32, QCOLS);

  v8f acc_o[8];
  float mstat[8], lsum[8];
#pragma unroll
  for (int j = 0; j < 8; ++j) {
#pragma unroll
    for (int e = 0; e < 8; ++e) acc_o[j][e] = 0.0f;
    mstat[j] = -3.0e38f;
    lsum[j]  = 0.0f;
  }

  const int nkv = (q0 >> 6) + 2;   // causal: keys < q0+128
  for (int t = 0; t < nkv; ++t) {
    const int kv0 = t * 64;
    __syncthreads();   // previous tile fully consumed
    // Stage K (64x128, direct copy) and V transposed (128x64).
    {
      const int c8  = (tid & 15) * 8;   // d-chunk (8 bf16 = 16B)
      const int k0r = tid >> 4;         // 0..15
#pragma unroll
      for (int i = 0; i < 4; ++i) {
        const int key = k0r + 16 * i;
        const size_t src =
            (size_t)(b * SEQ + kv0 + key) * KVCOLS + (size_t)kvh * HDIM + c8;
#ifdef USE_ASYNC_LDS
        __builtin_amdgcn_global_load_async_to_lds_b128(
            (gi4_t*)(unsigned long long)(Kc + src),
            (si4_t*)(unsigned long long)&Ks[key][c8], 0, 0);
#else
        *(BF8v*)&Ks[key][c8] = *reinterpret_cast<const BF8v*>(Kc + src);
#endif
        const BF8v vv = *reinterpret_cast<const BF8v*>(Vc + src);
#pragma unroll
        for (int j = 0; j < 8; ++j) Vs[c8 + j][key] = vv.v[j];
      }
    }
#ifdef USE_ASYNC_LDS
    __builtin_amdgcn_s_wait_asynccnt(0);
#endif
    __syncthreads();

    // S = scale * Q K^T (+ causal mask): 4 key-tiles x 4 k-steps of WMMA.
    v8f s[4];
#pragma unroll
    for (int nt = 0; nt < 4; ++nt)
#pragma unroll
      for (int e = 0; e < 8; ++e) s[nt][e] = 0.0f;
#pragma unroll
    for (int ks = 0; ks < 4; ++ks)
#pragma unroll
      for (int nt = 0; nt < 4; ++nt) {
        const v16bf bk = load_frag16(&Ks[nt * 16][ks * 32], LDK);
        s[nt] = wmma_bf16(aq[ks], bk, s[nt]);
      }

    // Online softmax per row (row m = r + 8*half lives in the 16-lane half).
#pragma unroll
    for (int r = 0; r < 8; ++r) {
      const int qi = q0 + wave * 16 + r + 8 * half;
      float mrow = mstat[r];
#pragma unroll
      for (int nt = 0; nt < 4; ++nt) {
        const int key = kv0 + nt * 16 + nl;
        float v = s[nt][r] * ATTN_SCALE + (key > qi ? NEG_BIG : 0.0f);
        s[nt][r] = v;
        mrow = fmaxf(mrow, v);
      }
#pragma unroll
      for (int off = 8; off >= 1; off >>= 1)
        mrow = fmaxf(mrow, __shfl_xor(mrow, off));
      const float alpha = __expf(mstat[r] - mrow);
      mstat[r] = mrow;
      float rsum = 0.0f;
#pragma unroll
      for (int nt = 0; nt < 4; ++nt) {
        const float p = __expf(s[nt][r] - mrow);
        s[nt][r] = p;
        rsum += p;
      }
#pragma unroll
      for (int off = 8; off >= 1; off >>= 1) rsum += __shfl_xor(rsum, off);
      lsum[r] = lsum[r] * alpha + rsum;
#pragma unroll
      for (int j = 0; j < 8; ++j) acc_o[j][r] *= alpha;
      // Spill P row chunk to wave-private LDS (C layout -> memory).
#pragma unroll
      for (int nt = 0; nt < 4; ++nt)
        Ps[wave][r + 8 * half][nt * 16 + nl] = (__bf16)s[nt][r];
    }

    // O += P V : 2 k-steps over 64 keys x 8 d-tiles.
#pragma unroll
    for (int ks = 0; ks < 2; ++ks) {
      const v16bf ap = load_frag16(&Ps[wave][0][ks * 32], LDP);
#pragma unroll
      for (int j = 0; j < 8; ++j) {
        const v16bf bv = load_frag16(&Vs[j * 16][ks * 32], LDV);
        acc_o[j] = wmma_bf16(ap, bv, acc_o[j]);
      }
    }
  }

  // Normalize and store (row = b*S + q, col = h*HD + d), bf16 scratch.
#pragma unroll
  for (int j = 0; j < 8; ++j)
#pragma unroll
    for (int r = 0; r < 8; ++r) {
      const int row = b * SEQ + q0 + wave * 16 + r + 8 * half;
      const int col = h * HDIM + j * 16 + nl;
      O[(size_t)row * QCOLS + col] = (__bf16)(acc_o[j][r] / lsum[r]);
    }
}

// ---------------------------------------------------------------------------
// Launch: Q/K/V GEMMs -> RoPE -> flash attention -> output projection.
// ---------------------------------------------------------------------------
extern "C" void kernel_launch(void* const* d_in, const int* in_sizes, int n_in,
                              void* d_out, int out_size, void* d_ws, size_t ws_size,
                              hipStream_t stream) {
  (void)in_sizes; (void)n_in; (void)out_size; (void)ws_size;
  const float* hs  = (const float*)d_in[0];
  const float* cs  = (const float*)d_in[1];
  const float* sn  = (const float*)d_in[2];
  // d_in[3] = attention_mask (causal; computed analytically in-kernel)
  const float* Wq  = (const float*)d_in[4];
  const float* Wk  = (const float*)d_in[5];
  const float* Wv  = (const float*)d_in[6];
  const float* Wo  = (const float*)d_in[7];
  float* out = (float*)d_out;

  __bf16* Qb = (__bf16*)d_ws;
  __bf16* Kb = Qb + (size_t)ROWS * QCOLS;    // +32MB
  __bf16* Vb = Kb + (size_t)ROWS * KVCOLS;   // +8MB
  __bf16* Ab = Vb + (size_t)ROWS * KVCOLS;   // +8MB (attn out, +32MB)

  // Projections (f32 in, bf16 out).
  gemm_wmma_kernel<float, __bf16><<<dim3(QCOLS / BN, ROWS / BM), 256, 0, stream>>>(
      hs, Wq, Qb, ROWS, QCOLS, DMODEL);
  gemm_wmma_kernel<float, __bf16><<<dim3(KVCOLS / BN, ROWS / BM), 256, 0, stream>>>(
      hs, Wk, Kb, ROWS, KVCOLS, DMODEL);
  gemm_wmma_kernel<float, __bf16><<<dim3(KVCOLS / BN, ROWS / BM), 256, 0, stream>>>(
      hs, Wv, Vb, ROWS, KVCOLS, DMODEL);

  // RoPE on Q and K.
  {
    const long long nq = (long long)ROWS * NHEADS * (HDIM / 2);
    rope_kernel<<<(unsigned)((nq + 255) / 256), 256, 0, stream>>>(Qb, cs, sn, NHEADS);
    const long long nk = (long long)ROWS * NKVH * (HDIM / 2);
    rope_kernel<<<(unsigned)((nk + 255) / 256), 256, 0, stream>>>(Kb, cs, sn, NKVH);
  }

  // Flash attention.
  flash_attn_kernel<<<dim3(SEQ / 128, BATCH * NHEADS), 256, 0, stream>>>(
      Qb, Kb, Vb, Ab);

  // Output projection (bf16 in, f32 out).
  gemm_wmma_kernel<__bf16, float><<<dim3(DMODEL / BN, ROWS / BM), 256, 0, stream>>>(
      Ab, Wo, out, ROWS, DMODEL, QCOLS);
}
